// _NonLocalBlockND_39668317946056
// MI455X (gfx1250) — matrix operand order repaired and checked by reading, not compiled, not run on hardware
//
#include <hip/hip_runtime.h>
#include <hip/hip_bf16.h>

// ---------------------------------------------------------------------------
// NonLocal (HiGCIN) block on MI455X / gfx1250. All GEMMs via
// v_wmma_f32_16x16x32_f16 (wave32), NT-form operands (both K-contiguous) so
// each fragment is two contiguous 16B loads per lane (ISA 7.12.2 layouts).
// Register-blocked: each wave computes an MTxNT grid of 16x16 tiles so A/B
// fragments are reused MT/NT times from registers.
// ---------------------------------------------------------------------------

#define B_   128
#define C_   1024
#define IC_  512
#define T_   20
#define S_   12
#define N_   240           // T*S
#define NP_  256           // N padded to multiple of 32 for K-dim use
#define M3_  1536          // 3*IC stacked projection rows (theta|phi|g)
#define BN_EPS 1e-5f

typedef __attribute__((ext_vector_type(16))) _Float16 v16h;
typedef __attribute__((ext_vector_type(8)))  _Float16 v8h;
typedef __attribute__((ext_vector_type(8)))  float    v8f;

// ---- WMMA fragment loaders (ISA 7.12.2, 16-bit A 16x32 / B 32x16) ---------
__device__ __forceinline__ v16h load_frag_a(const _Float16* A, int lda,
                                            int m0, int k0, int lane) {
  const int m = m0 + (lane & 15);
  const int h = lane >> 4;
  union { v16h v; v8h p[2]; } u;
  const _Float16* base = A + (size_t)m * lda + k0;
  u.p[0] = *(const v8h*)(base + 8 * h);        // K = k0 + 8h .. +7
  u.p[1] = *(const v8h*)(base + 16 + 8 * h);   // K = k0 + 16 + 8h .. +7
  return u.v;
}

__device__ __forceinline__ v16h load_frag_b(const _Float16* Bt, int ldb,
                                            int n0, int k0, int lane) {
  const int n = n0 + (lane & 15);
  const int h = lane >> 4;
  union { v16h v; v8h p[2]; } u;
  const _Float16* base = Bt + (size_t)n * ldb + k0;
  u.p[0] = *(const v8h*)(base + 8 * h);        // K = k0 + 8h .. +7
  u.p[1] = *(const v8h*)(base + 16 + 8 * h);   // K = k0 + 16 + 8h .. +7
  return u.v;
}

// Register-blocked NT GEMM: acc[mt][nt] = sum_k A[m0+16mt+.,k] * Bt[n0+16nt+.,k]
template <int MT, int NT>
__device__ __forceinline__ void gemm_tiles(const _Float16* __restrict__ A, int lda,
                                           const _Float16* __restrict__ Bt, int ldb,
                                           int K, int m0, int n0, int lane,
                                           v8f (&acc)[MT][NT]) {
#pragma unroll
  for (int i = 0; i < MT; ++i)
#pragma unroll
    for (int j = 0; j < NT; ++j) acc[i][j] = (v8f){};

#pragma unroll 2
  for (int k0 = 0; k0 < K; k0 += 32) {
    v16h a[MT], b[NT];
#pragma unroll
    for (int i = 0; i < MT; ++i) a[i] = load_frag_a(A, lda, m0 + 16 * i, k0, lane);
#pragma unroll
    for (int j = 0; j < NT; ++j) b[j] = load_frag_b(Bt, ldb, n0 + 16 * j, k0, lane);
#pragma unroll
    for (int i = 0; i < MT; ++i)
#pragma unroll
      for (int j = 0; j < NT; ++j)
        acc[i][j] = __builtin_amdgcn_wmma_f32_16x16x32_f16(
            false, a[i], false, b[j], (short)0, acc[i][j], false, false);
  }
}

// ---- P1: x (B,C,N) f32 -> xT (B,N,C) f16, LDS-tiled transpose --------------
__global__ void xpose_kernel(const float* __restrict__ x,
                             _Float16* __restrict__ xT) {
  __shared__ float tile[16][17];
  const int b  = blockIdx.z;
  const int n0 = blockIdx.x * 16;   // 15 blocks
  const int c0 = blockIdx.y * 16;   // 64 blocks
  const int tx = threadIdx.x, ty = threadIdx.y;
  tile[ty][tx] = x[((size_t)b * C_ + (c0 + ty)) * N_ + (n0 + tx)];
  __syncthreads();
  xT[((size_t)b * N_ + (n0 + ty)) * C_ + (c0 + tx)] = (_Float16)tile[tx][ty];
}

// ---- P2: convert weights to f16, zero K-pad columns of fh / gN -------------
__global__ void prep_kernel(const float* __restrict__ w_theta,
                            const float* __restrict__ w_phi,
                            const float* __restrict__ w_g,
                            const float* __restrict__ w_W,
                            _Float16* __restrict__ w3h,   // (1536,1024)
                            _Float16* __restrict__ wWh,   // (1024,512)
                            _Float16* __restrict__ fh,    // (B,240,256)
                            _Float16* __restrict__ gN) {  // (B,512,256)
  const size_t NW3 = (size_t)M3_ * C_;
  const size_t NWW = (size_t)C_ * IC_;
  const size_t NFP = (size_t)B_ * N_ * (NP_ - N_);
  const size_t NGP = (size_t)B_ * IC_ * (NP_ - N_);
  const size_t total = NW3 + NWW + NFP + NGP;
  for (size_t i = (size_t)blockIdx.x * blockDim.x + threadIdx.x; i < total;
       i += (size_t)gridDim.x * blockDim.x) {
    if (i < NW3) {
      size_t r = i >> 10, c = i & 1023;
      float v = (r < IC_) ? w_theta[r * C_ + c]
              : (r < 2 * IC_) ? w_phi[(r - IC_) * C_ + c]
                              : w_g[(r - 2 * IC_) * C_ + c];
      w3h[i] = (_Float16)v;
    } else if (i < NW3 + NWW) {
      size_t j = i - NW3;
      wWh[j] = (_Float16)w_W[j];
    } else if (i < NW3 + NWW + NFP) {
      size_t j = i - NW3 - NWW;
      fh[(j >> 4) * NP_ + N_ + (j & 15)] = (_Float16)0.f;
    } else {
      size_t j = i - NW3 - NWW - NFP;
      gN[(j >> 4) * NP_ + N_ + (j & 15)] = (_Float16)0.f;
    }
  }
}

// ---- K1: projections. D[o,n] = w3h[o,:] . xT[b,n,:] + bias ----------------
// Wave tile 32x48. rows 0..511 -> thetaT (B,N,IC); 512..1023 -> phiT;
// 1024..1535 -> gN (B,IC,NP) with N contiguous.
__global__ void proj_kernel(const _Float16* __restrict__ w3h,
                            const float* __restrict__ b_theta,
                            const float* __restrict__ b_phi,
                            const float* __restrict__ b_g,
                            const _Float16* __restrict__ xT,
                            _Float16* __restrict__ thetaT,
                            _Float16* __restrict__ phiT,
                            _Float16* __restrict__ gN) {
  const int lane = threadIdx.x & 31;
  const int tile = blockIdx.x * 8 + (threadIdx.x >> 5);
  const int tilesN = N_ / 48;                       // 5
  if (tile >= (M3_ / 32) * tilesN) return;          // 48*5=240, wave-uniform
  const int m0 = (tile / tilesN) * 32;
  const int n0 = (tile % tilesN) * 48;
  const int b = blockIdx.z;

  v8f acc[2][3];
  gemm_tiles<2, 3>(w3h, C_, xT + (size_t)b * N_ * C_, C_, C_, m0, n0, lane, acc);

  const int lane15 = lane & 15, h = lane >> 4;
#pragma unroll
  for (int mt = 0; mt < 2; ++mt) {
    const int ob = m0 + mt * 16 + 8 * h;
#pragma unroll
    for (int nt = 0; nt < 3; ++nt) {
      const int n = n0 + nt * 16 + lane15;
      if (ob < IC_) {
        union { v8h v; _Float16 e[8]; } s;
#pragma unroll
        for (int r = 0; r < 8; ++r) s.e[r] = (_Float16)(acc[mt][nt][r] + b_theta[ob + r]);
        *(v8h*)(thetaT + ((size_t)b * N_ + n) * IC_ + ob) = s.v;
      } else if (ob < 2 * IC_) {
        union { v8h v; _Float16 e[8]; } s;
#pragma unroll
        for (int r = 0; r < 8; ++r) s.e[r] = (_Float16)(acc[mt][nt][r] + b_phi[ob - IC_ + r]);
        *(v8h*)(phiT + ((size_t)b * N_ + n) * IC_ + (ob - IC_)) = s.v;
      } else {
        const int icb = ob - 2 * IC_;
#pragma unroll
        for (int r = 0; r < 8; ++r)
          gN[((size_t)b * IC_ + icb + r) * NP_ + n] = (_Float16)(acc[mt][nt][r] + b_g[icb + r]);
      }
    }
  }
}

// ---- K2: affinity. f[n,m] = (thetaT[b,n,:] . phiT[b,m,:]) * relmap[n,m] ----
// Wave tile 48x48.
__global__ void aff_kernel(const _Float16* __restrict__ thetaT,
                           const _Float16* __restrict__ phiT,
                           const float* __restrict__ relmap,
                           float* __restrict__ fOut,      // (B,N,N) f32
                           _Float16* __restrict__ fh) {   // (B,N,NP) f16
  const int lane = threadIdx.x & 31;
  const int tile = blockIdx.x * 8 + (threadIdx.x >> 5);
  const int tilesN = N_ / 48;                       // 5
  if (tile >= tilesN * tilesN) return;              // 25, wave-uniform
  const int m0 = (tile / tilesN) * 48;
  const int n0 = (tile % tilesN) * 48;
  const int b = blockIdx.z;

  v8f acc[3][3];
  gemm_tiles<3, 3>(thetaT + (size_t)b * N_ * IC_, IC_,
                   phiT + (size_t)b * N_ * IC_, IC_, IC_, m0, n0, lane, acc);

  const int lane15 = lane & 15, h = lane >> 4;
#pragma unroll
  for (int mt = 0; mt < 3; ++mt) {
    const int nbase = m0 + mt * 16 + 8 * h;
#pragma unroll
    for (int nt = 0; nt < 3; ++nt) {
      const int m = n0 + nt * 16 + lane15;
#pragma unroll
      for (int r = 0; r < 8; ++r) {
        const int n = nbase + r;
        const float v = acc[mt][nt][r] * relmap[(size_t)n * N_ + m];
        fOut[((size_t)b * N_ + n) * N_ + m] = v;
        fh[((size_t)b * N_ + n) * NP_ + m] = (_Float16)v;
      }
    }
  }
}

// ---- P3: nnz of f[0,0,:] -> recip = 1/nnz ---------------------------------
__global__ void nnz_kernel(const float* __restrict__ fOut,
                           float* __restrict__ recip) {
  __shared__ int cnt;
  if (threadIdx.x == 0) cnt = 0;
  __syncthreads();
  if (threadIdx.x < N_ && fOut[threadIdx.x] != 0.f) atomicAdd(&cnt, 1);
  __syncthreads();
  if (threadIdx.x == 0) recip[0] = 1.0f / (float)(cnt > 0 ? cnt : 1);
}

// ---- K3: aggregate. y[n,ic] = recip * sum_m fh[b,n,m] * gN[b,ic,m] --------
// Wave tile 48x32.
__global__ void apply_kernel(const _Float16* __restrict__ fh,
                             const _Float16* __restrict__ gN,
                             const float* __restrict__ recip,
                             _Float16* __restrict__ yT) {  // (B,N,IC)
  const int lane = threadIdx.x & 31;
  const int tile = blockIdx.x * 8 + (threadIdx.x >> 5);
  const int tilesN = IC_ / 32;                       // 16
  if (tile >= (N_ / 48) * tilesN) return;            // 80, wave-uniform
  const int m0 = (tile / tilesN) * 48;
  const int n0 = (tile % tilesN) * 32;
  const int b = blockIdx.z;

  v8f acc[3][2];
  gemm_tiles<3, 2>(fh + (size_t)b * N_ * NP_, NP_,
                   gN + (size_t)b * IC_ * NP_, NP_, NP_, m0, n0, lane, acc);

  const float rc = recip[0];
  const int lane15 = lane & 15, h = lane >> 4;
#pragma unroll
  for (int mt = 0; mt < 3; ++mt) {
    const int nbase = m0 + mt * 16 + 8 * h;
#pragma unroll
    for (int nt = 0; nt < 2; ++nt) {
      const int ic = n0 + nt * 16 + lane15;
#pragma unroll
      for (int r = 0; r < 8; ++r)
        yT[((size_t)b * N_ + nbase + r) * IC_ + ic] = (_Float16)(acc[mt][nt][r] * rc);
    }
  }
}

// ---- K4: output proj + eval BN + residual. Wave tile 32x48 ----------------
__global__ void outp_kernel(const _Float16* __restrict__ wWh,
                            const _Float16* __restrict__ yT,
                            const float* __restrict__ b_W,
                            const float* __restrict__ gamma,
                            const float* __restrict__ beta,
                            const float* __restrict__ run_mean,
                            const float* __restrict__ run_var,
                            const float* __restrict__ x,
                            float* __restrict__ zOut) {
  const int lane = threadIdx.x & 31;
  const int tile = blockIdx.x * 8 + (threadIdx.x >> 5);
  const int tilesN = N_ / 48;                        // 5
  if (tile >= (C_ / 32) * tilesN) return;            // 160, wave-uniform
  const int m0 = (tile / tilesN) * 32;
  const int n0 = (tile % tilesN) * 48;
  const int b = blockIdx.z;

  v8f acc[2][3];
  gemm_tiles<2, 3>(wWh, IC_, yT + (size_t)b * N_ * IC_, IC_, IC_, m0, n0, lane, acc);

  const int lane15 = lane & 15, h = lane >> 4;
#pragma unroll
  for (int mt = 0; mt < 2; ++mt) {
    const int cbase = m0 + mt * 16 + 8 * h;
#pragma unroll
    for (int nt = 0; nt < 3; ++nt) {
      const int n = n0 + nt * 16 + lane15;
#pragma unroll
      for (int r = 0; r < 8; ++r) {
        const int c = cbase + r;
        const float inv = gamma[c] * rsqrtf(run_var[c] + BN_EPS);
        const size_t idx = ((size_t)b * C_ + c) * N_ + n;
        zOut[idx] = (acc[mt][nt][r] + b_W[c] - run_mean[c]) * inv + beta[c] + x[idx];
      }
    }
  }
}

// ---------------------------------------------------------------------------
extern "C" void kernel_launch(void* const* d_in, const int* in_sizes, int n_in,
                              void* d_out, int out_size, void* d_ws, size_t ws_size,
                              hipStream_t stream) {
  (void)in_sizes; (void)n_in; (void)out_size; (void)ws_size;

  const float* x       = (const float*)d_in[0];
  const float* w_g     = (const float*)d_in[1];
  const float* b_g     = (const float*)d_in[2];
  const float* w_theta = (const float*)d_in[3];
  const float* b_theta = (const float*)d_in[4];
  const float* w_phi   = (const float*)d_in[5];
  const float* b_phi   = (const float*)d_in[6];
  const float* w_W     = (const float*)d_in[7];
  const float* b_W     = (const float*)d_in[8];
  const float* gamma   = (const float*)d_in[9];
  const float* beta    = (const float*)d_in[10];
  const float* rmean   = (const float*)d_in[11];
  const float* rvar    = (const float*)d_in[12];
  const float* relmap  = (const float*)d_in[13];

  float* zOut = (float*)d_out;                       // (B,C,T,S)
  float* fOut = zOut + (size_t)B_ * C_ * N_;         // (B,N,N)

  // workspace layout (f16 staging, ~207 MB; recip scalar first)
  float*    recip  = (float*)d_ws;
  _Float16* wsH    = (_Float16*)((char*)d_ws + 16);
  _Float16* xT     = wsH;                                   // B*N*C
  _Float16* thetaT = xT     + (size_t)B_ * N_ * C_;         // B*N*IC
  _Float16* phiT   = thetaT + (size_t)B_ * N_ * IC_;        // B*N*IC
  _Float16* gN     = phiT   + (size_t)B_ * N_ * IC_;        // B*IC*NP
  _Float16* fh     = gN     + (size_t)B_ * IC_ * NP_;       // B*N*NP
  _Float16* yT     = fh     + (size_t)B_ * N_ * NP_;        // B*N*IC
  _Float16* w3h    = yT     + (size_t)B_ * N_ * IC_;        // M3*C
  _Float16* wWh    = w3h    + (size_t)M3_ * C_;             // C*IC

  // P1: transpose x -> xT (f16)
  xpose_kernel<<<dim3(N_ / 16, C_ / 16, B_), dim3(16, 16), 0, stream>>>(x, xT);

  // P2: weight conversion + K-pad zeroing
  prep_kernel<<<14208, 256, 0, stream>>>(w_theta, w_phi, w_g, w_W,
                                         w3h, wWh, fh, gN);

  // K1: projections (1536 x 1024 x 240 per batch), 32x48 wave tiles
  proj_kernel<<<dim3((M3_ / 32) * (N_ / 48) / 8, 1, B_), 256, 0, stream>>>(
      w3h, b_theta, b_phi, b_g, xT, thetaT, phiT, gN);

  // K2: masked affinity (240 x 512 x 240 per batch), 48x48 wave tiles
  aff_kernel<<<dim3(((N_ / 48) * (N_ / 48) + 7) / 8, 1, B_), 256, 0, stream>>>(
      thetaT, phiT, relmap, fOut, fh);

  // P3: normalizer
  nnz_kernel<<<1, 256, 0, stream>>>(fOut, recip);

  // K3: aggregation (240 x 256 x 512 per batch), 48x32 wave tiles
  apply_kernel<<<dim3((N_ / 48) * (IC_ / 32) / 8, 1, B_), 256, 0, stream>>>(
      fh, gN, recip, yT);

  // K4: output projection + BN + residual (1024 x 512 x 240 per batch), 32x48
  outp_kernel<<<dim3((C_ / 32) * (N_ / 48) / 8, 1, B_), 256, 0, stream>>>(
      wWh, yT, b_W, gamma, beta, rmean, rvar, x, zOut);
}
